// GraphEventAttentionModule_84370337563175
// MI455X (gfx1250) — matrix operand order repaired and checked by reading, not compiled, not run on hardware
//
#include <hip/hip_runtime.h>
#include <hip/hip_bf16.h>
#include <cstdint>

typedef __attribute__((ext_vector_type(16))) _Float16     v16h;
typedef __attribute__((ext_vector_type(8)))  _Float16     v8h;
typedef __attribute__((ext_vector_type(8)))  float        v8f;
typedef __attribute__((ext_vector_type(4)))  unsigned int u32x4;
typedef __attribute__((ext_vector_type(8)))  int          i32x8;
typedef __attribute__((ext_vector_type(4)))  int          i32x4;

#define FDIM    512
#define NHEADS  4
#define SLEN    200
#define NBATCH  4
#define NEVENTS 8
#define NROWS   (NBATCH * SLEN)      // 800
#define CWIDE   (NHEADS * FDIM)     // 2048
#define KPAD    224                  // src dim padded to 7*32
#define AROWS   256                  // rows reserved per (b,h) attention block
#define ARSTR   40                   // padded LDS row stride (halfs) for raw A tile
#define NEGINF  (-3.402823466e38f)

// Tensor Data Mover availability (guarded; falls back to manual copies)
#if defined(__has_builtin)
#  if __has_builtin(__builtin_amdgcn_tensor_load_to_lds) && \
      __has_builtin(__builtin_amdgcn_s_wait_tensorcnt)
#    define USE_TDM 1
#  endif
#endif
#ifndef USE_TDM
#  define USE_TDM 0
#endif
#if defined(__has_include)
#  if __has_include(<hip/amd_detail/amd_gfx1250_TDM.h>)
#    define TDM_SIX_ARG 1
#  endif
#endif
#ifndef TDM_SIX_ARG
#  define TDM_SIX_ARG 0
#endif

// ---------------------------------------------------------------------------
// WMMA GEMM: C[z] = A[z] (MxK) * B[z] (KxN), f32 accumulate, f16 math.
// AF16=false: A is f32, staged (with cvt) into fragment-major LDS.
// AF16=true : A is f16 (attention matrix), DMA'd by the TDM into a raw
//             row-major LDS tile with hardware row padding.
// B is always f32, staged with cvt into fragment-major LDS.
// Requirements honored by all call sites: K%32==0 for AF16=false; N%128==0.
// ---------------------------------------------------------------------------
template <bool AF16>
__global__ __launch_bounds__(256)
void gemm_wmma_kernel(const void* __restrict__ Abase,
                      const float* __restrict__ Bbase, float* __restrict__ Cbase,
                      int M, int N, int K, int lda, int ldb, int ldc, int zdiv,
                      long long sA0, long long sA1,
                      long long sB0, long long sB1,
                      long long sC0, long long sC1) {
  __shared__ __align__(32) _Float16 As[128 * ARSTR];   // raw (AF16) or frag-major (f32)
  __shared__ __align__(32) _Float16 Bs[8 * 32 * 16];   // 32 x 128, frag-major

  const int tid  = threadIdx.x;
  const int lane = tid & 31;
  const int wave = tid >> 5;
  const int grp  = tid >> 5;
  const int hsel = lane >> 4;
  const int r16  = lane & 15;

  const int z  = blockIdx.z;
  const int z0 = z / zdiv;
  const int z1 = z % zdiv;
  const long long aoff = (long long)z0 * sA0 + (long long)z1 * sA1;
  const float* B = Bbase + (long long)z0 * sB0 + (long long)z1 * sB1;
  float*       C = Cbase + (long long)z0 * sC0 + (long long)z1 * sC1;
  const float*    Af = (const float*)Abase;
  const _Float16* Ah = (const _Float16*)Abase;

  const int blockM = blockIdx.y * 128;
  const int blockN = blockIdx.x * 128;

  const int mgw = (wave & 1) * 4;     // 2 (M) x 4 (N) wave grid
  const int ngw = (wave >> 1) * 2;

  v8f acc[4][2];
#pragma unroll
  for (int i = 0; i < 4; ++i)
#pragma unroll
    for (int j = 0; j < 2; ++j)
#pragma unroll
      for (int p = 0; p < 8; ++p) acc[i][j][p] = 0.0f;

  for (int k0 = 0; k0 < K; k0 += 32) {
    __syncthreads();

    // ---------------- stage A tile ----------------
    if (AF16) {
      // 128 x 32 halfs, row stride lda in global; raw LDS layout, stride ARSTR
#if USE_TDM
      if (tid < 32) {
        const unsigned long long gaddr =
            (unsigned long long)(uintptr_t)(Ah + aoff + (long long)blockM * lda) +
            (unsigned long long)k0 * 2ull;
        u32x4 g0;
        g0[0] = 1u;                                    // count=1, no gather
        g0[1] = (unsigned)(uintptr_t)&As[0];           // LDS byte offset (low 32b)
        g0[2] = (unsigned)gaddr;
        g0[3] = ((unsigned)(gaddr >> 32) & 0x1FFFFFFu) | 0x80000000u; // type=2
        i32x8 g1;
        g1[0] = (1 << 16)      // data_size = 2B
              | (1 << 20)      // pad_enable
              | (3 << 22)      // pad_interval: every 16 DWORDs (one 32-half row)
              | (3 << 25);     // pad_amount: 4 DWORDs -> row stride 40 halfs
        g1[1] = (KPAD & 0xFFFF) << 16;                  // tensor_dim0 = 224
        g1[2] = ((KPAD >> 16) & 0xFFFF) | ((4096 & 0xFFFF) << 16);  // tensor_dim1
        g1[3] = ((4096 >> 16) & 0xFFFF) | (32 << 16);   // tile_dim0 = 32
        g1[4] = 128;                                    // tile_dim1 = 128, tile_dim2 = 0
        g1[5] = KPAD;                                   // tensor_dim0_stride = 224
        g1[6] = 0;
        g1[7] = 0;
        i32x4 gz; gz[0] = gz[1] = gz[2] = gz[3] = 0;
#if TDM_SIX_ARG
        i32x8 gz8;
        for (int q = 0; q < 8; ++q) gz8[q] = 0;
        __builtin_amdgcn_tensor_load_to_lds(g0, g1, gz, gz, gz8, 0);
#else
        __builtin_amdgcn_tensor_load_to_lds(g0, g1, gz, gz, 0);
#endif
      }
#else
      {
        const int row = tid >> 1, ch = tid & 1;         // 2 x 32B chunks per row
        const v16h v =
            *(const v16h*)&Ah[aoff + (long long)(blockM + row) * lda + k0 + ch * 16];
        *(v16h*)&As[row * ARSTR + ch * 16] = v;
      }
#endif
    } else {
      // 128 x 32 f32 -> f16, fragment-major LDS
      const int row = blockM + grp * 16 + r16;
      v16h hv;
      if (row < M) {
        const float* src = Af + aoff + (long long)row * lda + k0 + hsel * 8;
        const float4 f0 = *(const float4*)(src + 0);
        const float4 f1 = *(const float4*)(src + 4);
        const float4 f2 = *(const float4*)(src + 16);
        const float4 f3 = *(const float4*)(src + 20);
        hv[0]  = (_Float16)f0.x; hv[1]  = (_Float16)f0.y;
        hv[2]  = (_Float16)f0.z; hv[3]  = (_Float16)f0.w;
        hv[4]  = (_Float16)f1.x; hv[5]  = (_Float16)f1.y;
        hv[6]  = (_Float16)f1.z; hv[7]  = (_Float16)f1.w;
        hv[8]  = (_Float16)f2.x; hv[9]  = (_Float16)f2.y;
        hv[10] = (_Float16)f2.z; hv[11] = (_Float16)f2.w;
        hv[12] = (_Float16)f3.x; hv[13] = (_Float16)f3.y;
        hv[14] = (_Float16)f3.z; hv[15] = (_Float16)f3.w;
      } else {
#pragma unroll
        for (int e = 0; e < 16; ++e) hv[e] = (_Float16)0.0f;
      }
      *(v16h*)&As[(grp * 32 + lane) * 16] = hv;
    }

    // ---------------- stage B tile (32 x 128, frag-major) ----------------
    {
      const int col = blockN + grp * 16 + r16;
      v16h hv;
#pragma unroll
      for (int e = 0; e < 16; ++e) {
        const int k = k0 + ((e < 8) ? (hsel * 8 + e) : (16 + hsel * 8 + (e - 8)));
        hv[e] = (k < K) ? (_Float16)B[(long long)k * ldb + col] : (_Float16)0.0f;
      }
      *(v16h*)&Bs[(grp * 32 + lane) * 16] = hv;
    }

#if USE_TDM
    if (AF16 && tid < 32) __builtin_amdgcn_s_wait_tensorcnt(0);
#endif
    __syncthreads();

    // ---------------- fragments + WMMA ----------------
    v16h afrag[4], bfrag[2];
#pragma unroll
    for (int i = 0; i < 4; ++i) {
      if (AF16) {
        const int row = (mgw + i) * 16 + r16;
        const v8h lo = *(const v8h*)&As[row * ARSTR + hsel * 8];
        const v8h hi = *(const v8h*)&As[row * ARSTR + 16 + hsel * 8];
#pragma unroll
        for (int e = 0; e < 8; ++e) { afrag[i][e] = lo[e]; afrag[i][8 + e] = hi[e]; }
      } else {
        afrag[i] = *(const v16h*)&As[((mgw + i) * 32 + lane) * 16];
      }
    }
#pragma unroll
    for (int j = 0; j < 2; ++j)
      bfrag[j] = *(const v16h*)&Bs[((ngw + j) * 32 + lane) * 16];

#pragma unroll
    for (int i = 0; i < 4; ++i)
#pragma unroll
      for (int j = 0; j < 2; ++j)
        acc[i][j] = __builtin_amdgcn_wmma_f32_16x16x32_f16(
            false, afrag[i], false, bfrag[j], (short)0, acc[i][j], false, false);
  }

  // ---------------- epilogue ----------------
#pragma unroll
  for (int i = 0; i < 4; ++i) {
#pragma unroll
    for (int j = 0; j < 2; ++j) {
      const int col = blockN + (ngw + j) * 16 + r16;
#pragma unroll
      for (int p = 0; p < 8; ++p) {
        const int row = blockM + (mgw + i) * 16 + hsel * 8 + p;
        if (row < M && col < N) C[(long long)row * ldc + col] = acc[i][j][p];
      }
    }
  }
}

// ---------------------------------------------------------------------------
// s_src[r,h] = sum_f proj[r, h*F+f] * a_src[h,f]  (one wave per (r,h))
// ---------------------------------------------------------------------------
__global__ void sst_kernel(const float* __restrict__ proj,
                           const float* __restrict__ a_src,
                           const float* __restrict__ a_tgt,
                           float* __restrict__ s_src, float* __restrict__ s_tgt) {
  const int idx  = blockIdx.x * 8 + (threadIdx.x >> 5);
  const int lane = threadIdx.x & 31;
  if (idx >= NROWS * NHEADS) return;
  const int h = idx & (NHEADS - 1);
  const int r = idx >> 2;
  float vs = 0.0f, vt = 0.0f;
  for (int f = lane; f < FDIM; f += 32) {
    const float p = proj[(long long)r * CWIDE + h * FDIM + f];
    vs += p * a_src[h * FDIM + f];
    vt += p * a_tgt[h * FDIM + f];
  }
  for (int off = 16; off > 0; off >>= 1) {
    vs += __shfl_xor(vs, off, 32);
    vt += __shfl_xor(vt, off, 32);
  }
  if (lane == 0) { s_src[idx] = vs; s_tgt[idx] = vt; }
}

// ---------------------------------------------------------------------------
// Masked leaky-relu scores + softmax over sources (mask built on the fly).
// Output f16 rows, K padded to KPAD, AROWS rows reserved per (b,h).
// ---------------------------------------------------------------------------
__global__ void attn_softmax_kernel(const float* __restrict__ preds,
                                    const float* __restrict__ s_src,
                                    const float* __restrict__ s_tgt,
                                    _Float16* __restrict__ attn, int e) {
  const int idx  = blockIdx.x * 8 + (threadIdx.x >> 5);
  const int lane = threadIdx.x & 31;
  if (idx >= NBATCH * NHEADS * SLEN) return;
  const int t  = idx % SLEN;
  const int bh = idx / SLEN;
  const int h  = bh % NHEADS;
  const int b  = bh / NHEADS;

  const float st     = s_tgt[((b * SLEN + t) * NHEADS) + h];
  const bool  conf_t = preds[((b * SLEN + t) * NEVENTS) + e] >= 0.5f;

  float vals[7];
  float mx = NEGINF;
#pragma unroll
  for (int c = 0; c < 7; ++c) {
    const int s = lane + c * 32;
    float sc = NEGINF;
    if (s < SLEN) {
      const bool m = (s == t) || (s == t + 1) || (s + 1 == t) ||
                     (conf_t && preds[((b * SLEN + s) * NEVENTS) + e] >= 0.5f);
      if (m) {
        const float x = s_src[(b * SLEN + s) * NHEADS + h] + st;
        sc = (x > 0.0f) ? x : 0.2f * x;
      }
    }
    vals[c] = sc;
    mx = fmaxf(mx, sc);
  }
  for (int off = 16; off > 0; off >>= 1) mx = fmaxf(mx, __shfl_xor(mx, off, 32));
  float sum = 0.0f;
#pragma unroll
  for (int c = 0; c < 7; ++c) {
    const float ev = (vals[c] <= NEGINF) ? 0.0f : __expf(vals[c] - mx);
    vals[c] = ev;
    sum += ev;
  }
  for (int off = 16; off > 0; off >>= 1) sum += __shfl_xor(sum, off, 32);
  const float inv = 1.0f / sum;
  _Float16* row = attn + ((long long)bh * AROWS + t) * KPAD;
#pragma unroll
  for (int c = 0; c < 7; ++c) row[lane + c * 32] = (_Float16)(vals[c] * inv);
}

// ---------------------------------------------------------------------------
__global__ void epi0_kernel(const float* __restrict__ aout,
                            const float* __restrict__ x,
                            const float* __restrict__ bias,
                            float* __restrict__ x1) {
  const long long i = (long long)blockIdx.x * 256 + threadIdx.x;
  if (i >= (long long)NROWS * CWIDE) return;
  const int c = (int)(i % CWIDE);
  const long long r = i / CWIDE;
  const float v = aout[i] + x[r * FDIM + (c & (FDIM - 1))] + bias[c];
  x1[i] = (v > 0.0f) ? v : (__expf(v) - 1.0f);
}

__global__ void epi1_kernel(const float* __restrict__ aout,
                            const float* __restrict__ skipp,
                            const float* __restrict__ bias,
                            float* __restrict__ yacc) {
  const long long i = (long long)blockIdx.x * 256 + threadIdx.x;
  if (i >= (long long)NROWS * FDIM) return;
  const int f = (int)(i % FDIM);
  const long long r = i / FDIM;
  float v = 0.0f;
#pragma unroll
  for (int h = 0; h < NHEADS; ++h) {
    const long long idx = r * CWIDE + h * FDIM + f;
    v += aout[idx] + skipp[idx];
  }
  v = v * (1.0f / NHEADS) + bias[f];
  yacc[i] += v * (1.0f / NEVENTS);
}

__global__ void zero_kernel(float* __restrict__ p, long long n) {
  const long long i = (long long)blockIdx.x * 256 + threadIdx.x;
  if (i < n) p[i] = 0.0f;
}

__global__ void zero_h_kernel(_Float16* __restrict__ p, long long n) {
  const long long i = (long long)blockIdx.x * 256 + threadIdx.x;
  if (i < n) p[i] = (_Float16)0.0f;
}

// ---------------------------------------------------------------------------
// Host-side orchestration
// ---------------------------------------------------------------------------
namespace {

struct Ws {
  float* x;       // NROWS x CWIDE (module io uses first NROWS*FDIM, row stride FDIM)
  float* proj;    // NROWS x CWIDE
  float* skipp;   // NROWS x CWIDE
  float* aout;    // NROWS x CWIDE
  float* x1;      // NROWS x CWIDE
  float* yacc;    // NROWS x FDIM
  float* ssrc;    // NROWS x NHEADS
  float* stgt;    // NROWS x NHEADS
  _Float16* attn; // NBATCH*NHEADS x AROWS x KPAD
};

inline void gemm_plain(const float* A, const float* B, float* C,
                       int M, int N, int K, int lda, int ldb, int ldc,
                       hipStream_t st) {
  dim3 grid((N + 127) / 128, (M + 127) / 128, 1);
  gemm_wmma_kernel<false><<<grid, 256, 0, st>>>(A, B, C, M, N, K, lda, ldb, ldc,
                                                1, 0, 0, 0, 0, 0, 0);
}

inline void gemm_attn(const _Float16* attn, const float* proj, float* C,
                      hipStream_t st) {
  dim3 grid((FDIM + 127) / 128, (SLEN + 127) / 128, NBATCH * NHEADS);
  gemm_wmma_kernel<true><<<grid, 256, 0, st>>>(
      attn, proj, C, SLEN, FDIM, SLEN, KPAD, CWIDE, CWIDE, NHEADS,
      (long long)NHEADS * AROWS * KPAD, (long long)AROWS * KPAD,
      (long long)SLEN * CWIDE, (long long)FDIM,
      (long long)SLEN * CWIDE, (long long)FDIM);
}

void run_modality(const float* feats, const float* preds,
                  const float* const* prm, float* out, const Ws& w,
                  hipStream_t st) {
  hipMemcpyAsync(w.x, feats, sizeof(float) * NROWS * FDIM,
                 hipMemcpyDeviceToDevice, st);

  for (int m = 0; m < 2; ++m) {
    const float* W0  = prm[m * 9 + 0];
    const float* as0 = prm[m * 9 + 1];
    const float* at0 = prm[m * 9 + 2];
    const float* b0  = prm[m * 9 + 3];
    const float* W1  = prm[m * 9 + 4];
    const float* as1 = prm[m * 9 + 5];
    const float* at1 = prm[m * 9 + 6];
    const float* b1  = prm[m * 9 + 7];
    const float* Wsk = prm[m * 9 + 8];

    gemm_plain(w.x, W0, w.proj, NROWS, CWIDE, FDIM, FDIM, CWIDE, CWIDE, st);
    sst_kernel<<<400, 256, 0, st>>>(w.proj, as0, at0, w.ssrc, w.stgt);
    zero_kernel<<<(NROWS * FDIM + 255) / 256, 256, 0, st>>>(
        w.yacc, (long long)NROWS * FDIM);

    for (int e = 0; e < NEVENTS; ++e) {
      attn_softmax_kernel<<<400, 256, 0, st>>>(preds, w.ssrc, w.stgt, w.attn, e);
      gemm_attn(w.attn, w.proj, w.aout, st);
      epi0_kernel<<<(int)(((long long)NROWS * CWIDE + 255) / 256), 256, 0, st>>>(
          w.aout, w.x, b0, w.x1);

      gemm_plain(w.x1, W1, w.proj, NROWS, CWIDE, CWIDE, CWIDE, CWIDE, CWIDE, st);
      gemm_plain(w.x1, Wsk, w.skipp, NROWS, CWIDE, CWIDE, CWIDE, CWIDE, CWIDE, st);
      sst_kernel<<<400, 256, 0, st>>>(w.proj, as1, at1, w.ssrc, w.stgt);
      attn_softmax_kernel<<<400, 256, 0, st>>>(preds, w.ssrc, w.stgt, w.attn, e);
      gemm_attn(w.attn, w.proj, w.aout, st);
      epi1_kernel<<<(NROWS * FDIM + 255) / 256, 256, 0, st>>>(
          w.aout, w.skipp, b1, w.yacc);
    }
    hipMemcpyAsync(w.x, w.yacc, sizeof(float) * NROWS * FDIM,
                   hipMemcpyDeviceToDevice, st);
  }
  hipMemcpyAsync(out, w.x, sizeof(float) * NROWS * FDIM,
                 hipMemcpyDeviceToDevice, st);
}

} // namespace

extern "C" void kernel_launch(void* const* d_in, const int* in_sizes, int n_in,
                              void* d_out, int out_size, void* d_ws, size_t ws_size,
                              hipStream_t stream) {
  (void)in_sizes; (void)n_in; (void)out_size; (void)ws_size;

  const float* video_feats = (const float*)d_in[0];
  const float* audio_feats = (const float*)d_in[1];
  const float* video_preds = (const float*)d_in[2];
  const float* audio_preds = (const float*)d_in[3];

  // jax pytree flatten: sorted keys ("audio"<"video"; "W"<"a_src"<"a_tgt"<
  // "bias"<"skip", None skip dropped):
  const float* prmA[18];
  const float* prmV[18];
  for (int i = 0; i < 18; ++i) prmA[i] = (const float*)d_in[4 + i];
  for (int i = 0; i < 18; ++i) prmV[i] = (const float*)d_in[4 + 18 + i];

  Ws w;
  float* p = (float*)d_ws;
  w.x     = p; p += (long long)NROWS * CWIDE;
  w.proj  = p; p += (long long)NROWS * CWIDE;
  w.skipp = p; p += (long long)NROWS * CWIDE;
  w.aout  = p; p += (long long)NROWS * CWIDE;
  w.x1    = p; p += (long long)NROWS * CWIDE;
  w.yacc  = p; p += (long long)NROWS * FDIM;
  w.ssrc  = p; p += NROWS * NHEADS;
  w.stgt  = p; p += NROWS * NHEADS;
  w.attn  = (_Float16*)p;

  const long long attn_elems = (long long)NBATCH * NHEADS * AROWS * KPAD;
  zero_h_kernel<<<(int)((attn_elems + 255) / 256), 256, 0, stream>>>(w.attn,
                                                                     attn_elems);

  float* out = (float*)d_out;
  run_modality(video_feats, video_preds, prmV, out, w, stream);
  run_modality(audio_feats, audio_preds, prmA, out + (long long)NROWS * FDIM, w,
               stream);
}
